// MHA_4956392259613
// MI455X (gfx1250) — matrix operand order repaired and checked by reading, not compiled, not run on hardware
//
#include <hip/hip_runtime.h>

#define L_SEQ 1024
#define BATCH 4
#define DMODEL 1024
#define HNO 16
#define HSZ 64

typedef __attribute__((ext_vector_type(16))) __bf16 v16bf;
typedef __attribute__((ext_vector_type(8)))  float  v8f;

struct alignas(16) I4 { int x, y, z, w; };
struct alignas(16) F4 { float x, y, z, w; };
struct alignas(8)  US4 { unsigned short a, b, c, d; };

union V16 {
    v16bf v;
    I4 q[2];
    unsigned short s[16];
};

// fp32 -> bf16, round-to-nearest-even
__device__ __forceinline__ unsigned short f2bf(float f) {
    unsigned int u = __float_as_uint(f);
    u = u + 0x7FFFu + ((u >> 16) & 1u);
    return (unsigned short)(u >> 16);
}

__device__ __forceinline__ v8f wmma_bf16(const V16& a, const V16& b, v8f c) {
    return __builtin_amdgcn_wmma_f32_16x16x32_bf16(
        /*neg_a=*/false, a.v, /*neg_b=*/false, b.v,
        /*c_mod=*/(short)0, c, /*reuse_a=*/false, /*reuse_b=*/false);
}

// Scheduler pipelining hints: interleave VMEM-read groups (mask 0x20) with
// WMMA groups (mask 0x8, MFMA/WMMA class) inside a single scheduling region.
#if __has_builtin(__builtin_amdgcn_sched_group_barrier)
#define SGB(m, n) __builtin_amdgcn_sched_group_barrier((m), (n), 0)
#define SCHED_PIPE_QKV()                         \
    SGB(0x020, 5); SGB(0x008, 2);                \
    SGB(0x020, 5); SGB(0x008, 2);                \
    SGB(0x020, 5); SGB(0x008, 2);                \
    SGB(0x020, 5); SGB(0x008, 2);
#define SCHED_PIPE_AV()                          \
    SGB(0x020, 6); SGB(0x008, 1);                \
    SGB(0x020, 6); SGB(0x008, 1);
#else
#define SCHED_PIPE_QKV()
#define SCHED_PIPE_AV()
#endif

// ---------------------------------------------------------------------------
// TDM: stage one head's K slice (1024 x 64 bf16, 128 KB) into LDS.
// D# per CDNA5 ISA ch.8: group0 = {flags, lds_addr, global_addr, type=2},
// group1 = {data_size=2B, tensor_dim0=64, tensor_dim1=1024, tile 64x1024,
//           dim0_stride=64}; groups 2/3 zero (2D tensor).
// This toolchain exposes the 6-arg builtin:
//   (u32x4 g0, i32x8 g1, i32x4 g2, i32x4 g3, i32x8 extra, i32 cpol)
// ---------------------------------------------------------------------------
#if defined(__gfx1250__) && __has_builtin(__builtin_amdgcn_tensor_load_to_lds)
#define HAVE_TDM 1
typedef __attribute__((ext_vector_type(4))) unsigned int u32x4;
typedef __attribute__((ext_vector_type(8))) int i32x8;
typedef __attribute__((ext_vector_type(4))) int i32x4;

__device__ __forceinline__ void tdm_load_khead(const unsigned short* gsrc,
                                               unsigned int lds_byte_off) {
    unsigned long long ga = (unsigned long long)(uintptr_t)gsrc;
    u32x4 g0;
    g0.x = 1u;                                     // count=1, user-mode D#
    g0.y = lds_byte_off;                           // lds_addr
    g0.z = (unsigned int)ga;                       // global_addr[31:0]
    g0.w = (unsigned int)((ga >> 32) & 0x01FFFFFFu) | (2u << 30); // type=2
    i32x8 g1;
    g1[0] = (int)(1u << 16);        // data_size = 1 -> 2 bytes
    g1[1] = (int)(64u << 16);       // tensor_dim0 = 64  (bits 79:48, low16)
    g1[2] = (int)(1024u << 16);     // tensor_dim1 = 1024 (bits 111:80, low16)
    g1[3] = (int)(64u << 16);       // tile_dim0 = 64 (bits 127:112)
    g1[4] = 1024;                   // tile_dim1 = 1024 (bits 143:128)
    g1[5] = 64;                     // tensor_dim0_stride = 64
    g1[6] = 0;
    g1[7] = 0;
    i32x4 g2 = {0, 0, 0, 0};
    i32x4 g3 = {0, 0, 0, 0};
    i32x8 gx = {0, 0, 0, 0, 0, 0, 0, 0};
    __builtin_amdgcn_tensor_load_to_lds(g0, g1, g2, g3, gx, 0);
}
#endif

// ---------------------------------------------------------------------------
// Elementwise fp32 -> bf16 (vectorized x4)
// ---------------------------------------------------------------------------
__global__ void cvt_f32_bf16(const float* __restrict__ src,
                             unsigned short* __restrict__ dst, int n4) {
    int i = blockIdx.x * blockDim.x + threadIdx.x;
    if (i >= n4) return;
    F4 f = ((const F4*)src)[i];
    US4 r = { f2bf(f.x), f2bf(f.y), f2bf(f.z), f2bf(f.w) };
    ((US4*)dst)[i] = r;
}

// ---------------------------------------------------------------------------
// QKV projection: y = x @ W^T + b (M=4096, N=1024, K=1024) for q,k,v.
// One wave computes a 16x64 tile; K-loop of 64, double-buffered, last
// iteration peeled so the hot body is a single scheduling region.
// q,k stored bf16 head-split [bh][l][d]; v stored transposed [bh][d][l].
// ---------------------------------------------------------------------------
__global__ void __launch_bounds__(256) qkv_gemm(
    const unsigned short* __restrict__ xbf,
    const unsigned short* __restrict__ wqbf,
    const unsigned short* __restrict__ wkbf,
    const unsigned short* __restrict__ wvbf,
    const float* __restrict__ bq, const float* __restrict__ bk,
    const float* __restrict__ bv,
    unsigned short* __restrict__ qs,
    unsigned short* __restrict__ ks,
    unsigned short* __restrict__ vts) {
    int wid  = (blockIdx.x * blockDim.x + threadIdx.x) >> 5;
    int lane = threadIdx.x & 31;
    int mat  = wid >> 12;              // 0=q, 1=k, 2=v  (4096 waves per matrix)
    int rem  = wid & 4095;
    int m0   = (rem >> 4) << 4;        // 256 M-tiles of 16
    int n0   = (rem & 15) << 6;        // 16 N-groups of 64

    const unsigned short* wb = (mat == 0) ? wqbf : (mat == 1) ? wkbf : wvbf;
    const float* bias        = (mat == 0) ? bq   : (mat == 1) ? bk   : bv;

    int lhalf = lane >> 4;             // A, B: K-subblock select (same k order)
    int l15   = lane & 15;
    const unsigned short* arow = xbf + (size_t)(m0 + l15) * DMODEL;
    int akoff = lhalf * 8;
    int bkoff = lhalf * 8;

#define LOAD_A(dst, kk)                                        \
    dst.q[0] = *(const I4*)(arow + (kk) + akoff);              \
    dst.q[1] = *(const I4*)(arow + (kk) + 16 + akoff);
#define LOAD_B(dst, nt, kk) {                                               \
    const unsigned short* brow_ =                                           \
        wb + (size_t)(n0 + (nt) * 16 + l15) * DMODEL + (kk) + bkoff;        \
    dst.q[0] = *(const I4*)brow_;                                           \
    dst.q[1] = *(const I4*)(brow_ + 16); }

    v8f c[4] = { {}, {}, {}, {} };
    V16 aA, aB, bA[4], bB[4];

    LOAD_A(aA, 0);
#pragma unroll
    for (int nt = 0; nt < 4; ++nt) LOAD_B(bA[nt], nt, 0);

    int k0 = 0;
    for (; k0 < DMODEL - 64; k0 += 64) {
        LOAD_A(aB, k0 + 32);
#pragma unroll
        for (int nt = 0; nt < 4; ++nt) LOAD_B(bB[nt], nt, k0 + 32);
#pragma unroll
        for (int nt = 0; nt < 4; ++nt) c[nt] = wmma_bf16(aA, bA[nt], c[nt]);
        LOAD_A(aA, k0 + 64);
#pragma unroll
        for (int nt = 0; nt < 4; ++nt) LOAD_B(bA[nt], nt, k0 + 64);
#pragma unroll
        for (int nt = 0; nt < 4; ++nt) c[nt] = wmma_bf16(aB, bB[nt], c[nt]);
        SCHED_PIPE_QKV();
    }
    // peeled tail: k0 == DMODEL-64
    LOAD_A(aB, k0 + 32);
#pragma unroll
    for (int nt = 0; nt < 4; ++nt) LOAD_B(bB[nt], nt, k0 + 32);
#pragma unroll
    for (int nt = 0; nt < 4; ++nt) c[nt] = wmma_bf16(aA, bA[nt], c[nt]);
#pragma unroll
    for (int nt = 0; nt < 4; ++nt) c[nt] = wmma_bf16(aB, bB[nt], c[nt]);
#undef LOAD_A
#undef LOAD_B

#pragma unroll
    for (int nt = 0; nt < 4; ++nt) {
        int n = n0 + nt * 16 + l15;
        float bn = bias[n];
        int h = n >> 6, dd = n & 63;
#pragma unroll
        for (int r = 0; r < 8; ++r) {
            int m  = m0 + r + lhalf * 8;      // row of flattened (L*B, D)
            int li = m >> 2;                  // l  (B==4)
            int bi = m & 3;                   // batch
            int bhh = bi * HNO + h;
            unsigned short val = f2bf(c[nt][r] + bn);
            if (mat == 2)
                vts[(size_t)(bhh * HSZ + dd) * L_SEQ + li] = val;
            else if (mat == 0)
                qs[(size_t)(bhh * L_SEQ + li) * HSZ + dd] = val;
            else
                ks[(size_t)(bhh * L_SEQ + li) * HSZ + dd] = val;
        }
    }
}

// ---------------------------------------------------------------------------
// scores = q k^T / 8 + bias + mask, then row softmax.
// One block = (head bh, 16-row tile). Dynamic LDS: [0,64K) fp32 score rows,
// [64K,192K) the head's K slice staged once by the Tensor Data Mover.
// 4 waves x 256 cols each via WMMA (B operand from LDS), then an
// 8-lanes-per-row shuffle-reduced softmax, fp32 weights out.
// ---------------------------------------------------------------------------
__global__ void __launch_bounds__(128) attn_scores_softmax(
    const unsigned short* __restrict__ qs,
    const unsigned short* __restrict__ ks,
    const float* __restrict__ attn_bias,
    const float* __restrict__ mask,
    float* __restrict__ wout) {
    extern __shared__ char smem_dyn[];
    float* scores = (float*)smem_dyn;                         // 16 x 1024 fp32
    unsigned short* ldsk = (unsigned short*)(smem_dyn + 65536); // 1024 x 64 bf16

    int bh = blockIdx.x >> 6;
    int m0 = (blockIdx.x & 63) << 4;
    int w    = threadIdx.x >> 5;
    int lane = threadIdx.x & 31;
    int lhalf = lane >> 4, l15 = lane & 15;

    // Stage k[bh] (128 KB) into LDS.
#ifdef HAVE_TDM
    if (threadIdx.x < 32) {
        tdm_load_khead(ks + (size_t)bh * L_SEQ * HSZ, 65536u);
        __builtin_amdgcn_s_wait_tensorcnt(0);
    }
    __syncthreads();
#else
    {
        const I4* src = (const I4*)(ks + (size_t)bh * L_SEQ * HSZ);
        I4* dst = (I4*)ldsk;
        for (int i = threadIdx.x; i < 8192; i += 128) dst[i] = src[i];
        __syncthreads();
    }
#endif

    // A = q tile (16 x 64), loaded once from global
    const unsigned short* qrow = qs + (size_t)(bh * L_SEQ + m0 + l15) * HSZ;
    int akoff = lhalf * 8;
    V16 a0, a1;
    a0.q[0] = *(const I4*)(qrow + akoff);
    a0.q[1] = *(const I4*)(qrow + 16 + akoff);
    a1.q[0] = *(const I4*)(qrow + 32 + akoff);
    a1.q[1] = *(const I4*)(qrow + 48 + akoff);

    for (int nt = 0; nt < 16; ++nt) {
        int nc0 = (w << 8) + (nt << 4);
        const unsigned short* krow = ldsk + (nc0 + l15) * HSZ + lhalf * 8;
        V16 b0, b1;
        b0.q[0] = *(const I4*)(krow);
        b0.q[1] = *(const I4*)(krow + 16);
        b1.q[0] = *(const I4*)(krow + 32);
        b1.q[1] = *(const I4*)(krow + 48);
        v8f c = {};
        c = wmma_bf16(a0, b0, c);
        c = wmma_bf16(a1, b1, c);
        int col = nc0 + l15;
#pragma unroll
        for (int r = 0; r < 8; ++r) {
            int rl = r + lhalf * 8;
            int rg = m0 + rl;
            float v = c[r] * 0.125f
                    + attn_bias[(size_t)(bh * L_SEQ + rg) * L_SEQ + col]
                    + mask[(size_t)rg * L_SEQ + col];
            scores[rl * L_SEQ + col] = v;
        }
    }
    __syncthreads();

    // softmax: 8 lanes per row, rows 4w..4w+3 per wave
    int rl  = (w << 2) + (lane >> 3);
    int sub = lane & 7;
    float mx = -3.0e38f;
    for (int j = sub; j < L_SEQ; j += 8) mx = fmaxf(mx, scores[rl * L_SEQ + j]);
#pragma unroll
    for (int o = 1; o < 8; o <<= 1) mx = fmaxf(mx, __shfl_xor(mx, o, 32));
    float s = 0.f;
    for (int j = sub; j < L_SEQ; j += 8) {
        float e = __expf(scores[rl * L_SEQ + j] - mx);
        scores[rl * L_SEQ + j] = e;
        s += e;
    }
#pragma unroll
    for (int o = 1; o < 8; o <<= 1) s += __shfl_xor(s, o, 32);
    float inv = 1.0f / s;
    float* wrow = wout + (size_t)(bh * L_SEQ + m0 + rl) * L_SEQ;
    for (int j = sub; j < L_SEQ; j += 8) wrow[j] = scores[rl * L_SEQ + j] * inv;
}

// ---------------------------------------------------------------------------
// value = P @ v. A = fp32 weights -> bf16 on the fly, B = v^T bf16.
// One wave per 16x16 output tile, K = 1024 in steps of 32, double-buffered,
// last iteration peeled for a branch-free pipelined body.
// ---------------------------------------------------------------------------
__global__ void __launch_bounds__(128) attn_value(
    const float* __restrict__ weights,
    const unsigned short* __restrict__ vts,
    float* __restrict__ vout) {
    int bh = blockIdx.x >> 6;
    int m0 = (blockIdx.x & 63) << 4;
    int w    = threadIdx.x >> 5;
    int lane = threadIdx.x & 31;
    int lhalf = lane >> 4, l15 = lane & 15;
    int d0 = w << 4;

    const float* prow = weights + (size_t)(bh * L_SEQ + m0 + l15) * L_SEQ;
    int akoff = lhalf * 8;
    const unsigned short* vrow =
        vts + (size_t)(bh * HSZ + d0 + l15) * L_SEQ + lhalf * 8;

#define LOAD_AF(dst, kk)                                 \
    dst[0] = *(const F4*)(prow + (kk) + akoff);          \
    dst[1] = *(const F4*)(prow + (kk) + akoff + 4);      \
    dst[2] = *(const F4*)(prow + (kk) + 16 + akoff);     \
    dst[3] = *(const F4*)(prow + (kk) + 16 + akoff + 4);
#define LOAD_BV(dst, kk)                                 \
    dst.q[0] = *(const I4*)(vrow + (kk));                \
    dst.q[1] = *(const I4*)(vrow + (kk) + 16);
#define CVT16(a, f)                                      \
    a.s[0]  = f2bf(f[0].x); a.s[1]  = f2bf(f[0].y);      \
    a.s[2]  = f2bf(f[0].z); a.s[3]  = f2bf(f[0].w);      \
    a.s[4]  = f2bf(f[1].x); a.s[5]  = f2bf(f[1].y);      \
    a.s[6]  = f2bf(f[1].z); a.s[7]  = f2bf(f[1].w);      \
    a.s[8]  = f2bf(f[2].x); a.s[9]  = f2bf(f[2].y);      \
    a.s[10] = f2bf(f[2].z); a.s[11] = f2bf(f[2].w);      \
    a.s[12] = f2bf(f[3].x); a.s[13] = f2bf(f[3].y);      \
    a.s[14] = f2bf(f[3].z); a.s[15] = f2bf(f[3].w);

    v8f c = {};
    F4 fA[4], fB[4];
    V16 bA, bB;
    LOAD_AF(fA, 0);
    LOAD_BV(bA, 0);
    int k0 = 0;
    for (; k0 < L_SEQ - 64; k0 += 64) {
        LOAD_AF(fB, k0 + 32);
        LOAD_BV(bB, k0 + 32);
        V16 a0;
        CVT16(a0, fA);
        c = wmma_bf16(a0, bA, c);
        LOAD_AF(fA, k0 + 64);
        LOAD_BV(bA, k0 + 64);
        V16 a1;
        CVT16(a1, fB);
        c = wmma_bf16(a1, bB, c);
        SCHED_PIPE_AV();
    }
    // peeled tail: k0 == L_SEQ-64
    LOAD_AF(fB, k0 + 32);
    LOAD_BV(bB, k0 + 32);
    {
        V16 a0;
        CVT16(a0, fA);
        c = wmma_bf16(a0, bA, c);
        V16 a1;
        CVT16(a1, fB);
        c = wmma_bf16(a1, bB, c);
    }
#undef LOAD_AF
#undef LOAD_BV
#undef CVT16

    int bi = bh >> 4, h = bh & 15;
    int col = h * HSZ + d0 + l15;
#pragma unroll
    for (int r = 0; r < 8; ++r) {
        int lrow = m0 + r + lhalf * 8;
        vout[(size_t)(lrow * BATCH + bi) * DMODEL + col] = c[r];
    }
}

// ---------------------------------------------------------------------------
extern "C" void kernel_launch(void* const* d_in, const int* in_sizes, int n_in,
                              void* d_out, int out_size, void* d_ws, size_t ws_size,
                              hipStream_t stream) {
    const float* x         = (const float*)d_in[0];
    const float* attn_bias = (const float*)d_in[1];
    const float* mask      = (const float*)d_in[2];
    const float* Wq        = (const float*)d_in[3];
    const float* bq        = (const float*)d_in[4];
    const float* Wk        = (const float*)d_in[5];
    const float* bk        = (const float*)d_in[6];
    const float* Wv        = (const float*)d_in[7];
    const float* bv        = (const float*)d_in[8];

    float* out         = (float*)d_out;
    float* value_out   = out;                                    // L*B*D
    float* weights_out = out + (size_t)L_SEQ * BATCH * DMODEL;   // BH*L*L

    // workspace (bf16 staging), ~38 MB total
    unsigned short* xbf  = (unsigned short*)d_ws;   // 4194304
    unsigned short* wqbf = xbf  + 4194304;          // 1048576
    unsigned short* wkbf = wqbf + 1048576;
    unsigned short* wvbf = wkbf + 1048576;
    unsigned short* qs   = wvbf + 1048576;          // 4194304 [bh][l][d]
    unsigned short* ksb  = qs   + 4194304;          // 4194304 [bh][l][d]
    unsigned short* vts  = ksb  + 4194304;          // 4194304 [bh][d][l]

    cvt_f32_bf16<<<4096, 256, 0, stream>>>(x,  xbf,  1048576);
    cvt_f32_bf16<<<1024, 256, 0, stream>>>(Wq, wqbf, 262144);
    cvt_f32_bf16<<<1024, 256, 0, stream>>>(Wk, wkbf, 262144);
    cvt_f32_bf16<<<1024, 256, 0, stream>>>(Wv, wvbf, 262144);

    // 3 matrices * 256 m-tiles * 16 n-groups = 12288 waves / 8 per block
    qkv_gemm<<<1536, 256, 0, stream>>>(xbf, wqbf, wkbf, wvbf,
                                       bq, bk, bv, qs, ksb, vts);

    // 64 heads * 64 row-tiles; 192 KB dynamic LDS (scores + TDM-staged K)
    attn_scores_softmax<<<4096, 128, 196608, stream>>>(qs, ksb, attn_bias, mask,
                                                       weights_out);
    attn_value<<<4096, 128, 0, stream>>>(weights_out, vts, value_out);
}
